// WindowAttentionReverse_67001489818234
// MI455X (gfx1250) — matrix operand order repaired and checked by reading, not compiled, not run on hardware
//
#include <hip/hip_runtime.h>
#include <hip/hip_bf16.h>

typedef __attribute__((ext_vector_type(16))) _Float16 v16h;
typedef __attribute__((ext_vector_type(8)))  float    v8f;

#define Bn   256
#define Nn   210
#define Dm   192
#define Hh   6
#define HD   32
#define NT   14      // ceil(210/16)
#define NR   224     // padded row count (NT*16)
#define NWIN 64

__device__ __forceinline__ v8f wmma16(v16h a, v16h b, v8f c) {
  // D = A(16x32 f16) * B(32x16 f16) + C(16x16 f32)
  return __builtin_amdgcn_wmma_f32_16x16x32_f16(false, a, false, b, (short)0, c,
                                                false, false);
}

// Build A-frag from a row-major f16 row pointer.
// lane l holds row m=l%16, contiguous K = 2*kb+{0..15}, kb=(l/16)*8
// (matches contiguous B fragments).
__device__ __forceinline__ v16h fragA(const _Float16* row, int kb) {
  v16h a;
#pragma unroll
  for (int i = 0; i < 8; ++i) { a[i] = row[2 * kb + i]; a[8 + i] = row[2 * kb + 8 + i]; }
  return a;
}

union H4U2 { _Float16 h[4]; uint2 u; };
union U4H8 { uint4 u; _Float16 h[8]; };

// ---------------------------------------------------------------------------
// Kernel 0a: gather relative-position bias into [H][N][N] f32 (L2 resident)
// ---------------------------------------------------------------------------
__global__ void bias_kernel(const int* __restrict__ rel_idx,
                            const float* __restrict__ rpb,
                            float* __restrict__ biasws) {
  int i = blockIdx.x * blockDim.x + threadIdx.x;
  if (i >= Hh * Nn * Nn) return;
  int h = i / (Nn * Nn);
  int j = i - h * (Nn * Nn);
  biasws[i] = rpb[rel_idx[j] * Hh + h];
}

// ---------------------------------------------------------------------------
// Kernel 0b: combined (bias+mask) f16 table, rows padded to NR:
// CB[((w*H + h)*NR + m)*Nn + n] = bias[h][m][n] + mask[w][m][n]  (m<Nn else 0)
// -> attention adds become constant-offset b16 loads off one base per rowblock
// ---------------------------------------------------------------------------
__global__ void comb_kernel(const float* __restrict__ bias,
                            const float* __restrict__ mask,
                            _Float16* __restrict__ CB) {
  int i = blockIdx.x * blockDim.x + threadIdx.x;
  if (i >= NWIN * Hh * NR * Nn) return;
  int n = i % Nn;
  int t = i / Nn;
  int m = t % NR; t /= NR;
  int h = t % Hh;
  int w = t / Hh;
  float v = 0.f;
  if (m < Nn)
    v = bias[((size_t)h * Nn + m) * Nn + n] + mask[((size_t)w * Nn + m) * Nn + n];
  CB[i] = (_Float16)v;
}

// ---------------------------------------------------------------------------
// Kernel 0c: pack weight matrix (192 x cols, f32) into B-frag layout f16:
// out[(((tile*6+kc)*2+half)*16+ln)*16+e] = W[(kc*32+half*16+e)*cols + tile*16+ln]
// ---------------------------------------------------------------------------
__global__ void pack_w_kernel(const float* __restrict__ W, int cols,
                              _Float16* __restrict__ out) {
  int i = blockIdx.x * blockDim.x + threadIdx.x;
  if (i >= Dm * cols) return;
  int e    = i & 15;
  int ln   = (i >> 4) & 15;
  int half = (i >> 8) & 1;
  int kc   = (i >> 9) % 6;
  int tile = (i >> 9) / 6;
  int row = kc * 32 + half * 16 + e;
  int col = tile * 16 + ln;
  out[i] = (_Float16)W[(size_t)row * cols + col];
}

// ---------------------------------------------------------------------------
// Kernel 1: fused QKV projections (f32 in -> f16 Q/K/V in [B][H][NR][32])
// grid (NT, B), 256 threads = 8 waves; 72 output 16-col tiles, 9 per wave
// Pad rows (>=Nn) computed from zero inputs -> finite, stored unconditionally.
// ---------------------------------------------------------------------------
__global__ void qkv_kernel(const float* __restrict__ mov, const float* __restrict__ fix,
                           const float* __restrict__ dmov, const float* __restrict__ dfix,
                           const _Float16* __restrict__ Pmq, const float* __restrict__ mqb,
                           const _Float16* __restrict__ Pmkv, const float* __restrict__ mkvb,
                           const _Float16* __restrict__ Pfq, const float* __restrict__ fqb,
                           const _Float16* __restrict__ Pfkv, const float* __restrict__ fkvb,
                           _Float16* __restrict__ Qm, _Float16* __restrict__ Qf,
                           _Float16* __restrict__ Km, _Float16* __restrict__ Vm,
                           _Float16* __restrict__ Kf, _Float16* __restrict__ Vf) {
  const int rt = blockIdx.x, b = blockIdx.y;
  __shared__ _Float16 X[4][16 * Dm];
  // vectorized staging: 4 inputs x 16 rows x 48 float4 (guard only vs d_in)
  for (int i = threadIdx.x; i < 4 * 16 * 48; i += blockDim.x) {
    int which = i / (16 * 48), j = i - which * (16 * 48);
    int r = j / 48, c4 = (j - r * 48) * 4;
    int gr = rt * 16 + r;
    const float* sp = (which == 0) ? mov : (which == 1) ? fix : (which == 2) ? dmov : dfix;
    float4 f = make_float4(0.f, 0.f, 0.f, 0.f);
    if (gr < Nn) f = *(const float4*)(sp + ((size_t)b * Nn + gr) * Dm + c4);
    H4U2 p;
    p.h[0] = (_Float16)f.x; p.h[1] = (_Float16)f.y;
    p.h[2] = (_Float16)f.z; p.h[3] = (_Float16)f.w;
    *(uint2*)&X[which][r * Dm + c4] = p.u;
  }
  __syncthreads();

  const int lane = threadIdx.x & 31, wv = threadIdx.x >> 5;
  const int ln = lane & 15, half = lane >> 4, kb = half * 8;
  const float scale = 0.17677669529663687f;  // 32^-0.5

  for (int tt = 0; tt < 9; ++tt) {
    int t = wv * 9 + tt;
    int which, tl, dk;
    const _Float16* Wp;
    const float* bv;
    if (t < 12)      { which = 0; Wp = Pmq;  bv = mqb;  tl = t;      dk = 0; }
    else if (t < 24) { which = 1; Wp = Pfq;  bv = fqb;  tl = t - 12; dk = 1; }
    else if (t < 48) { which = 2; Wp = Pmkv; bv = mkvb; tl = t - 24; dk = 2; }
    else             { which = 3; Wp = Pfkv; bv = fkvb; tl = t - 48; dk = 3; }

    __builtin_prefetch(Wp + (size_t)tl * 6 * 512, 0, 1);
    v8f acc = {0.f, 0.f, 0.f, 0.f, 0.f, 0.f, 0.f, 0.f};
#pragma unroll
    for (int kc = 0; kc < 6; ++kc) {
      v16h a = fragA(&X[which][ln * Dm + kc * 32], kb);
      const v16h* wp = (const v16h*)(Wp + ((((size_t)tl * 6 + kc) * 2 + half) * 16 + ln) * 16);
      acc = wmma16(a, *wp, acc);
    }
    int c = tl * 16 + ln;
    float bb = bv[c];
    // destination base (padded NR rows -> unconditional stores)
    _Float16* Dp;
    int c2 = c;
    float mul = 1.0f;
    if (dk == 0)      { Dp = Qm; mul = scale; }
    else if (dk == 1) { Dp = Qf; mul = scale; }
    else {
      if (c2 < Dm) Dp = (dk == 2) ? Km : Kf;
      else { c2 -= Dm; Dp = (dk == 2) ? Vm : Vf; }
    }
    int h = c2 / HD;
    _Float16* dst = Dp + (((size_t)b * Hh + h) * NR + rt * 16 + half * 8) * HD + (c2 & 31);
#pragma unroll
    for (int r = 0; r < 8; ++r)
      dst[r * HD] = (_Float16)((acc[r] + bb) * mul);
  }
}

// ---------------------------------------------------------------------------
// Kernel 2: attention for one (b,h): S = Q*K^T + (bias+mask), softmax, P*V
// grid (H, B), 128 threads = 4 waves, each wave owns 16 query rows at a time
// ---------------------------------------------------------------------------
__global__ void attn_kernel(const _Float16* __restrict__ Q,   // [B][H][NR][32]
                            const _Float16* __restrict__ K,   // [B][H][NR][32]
                            const _Float16* __restrict__ V,   // [B][H][NR][32]
                            const _Float16* __restrict__ CB,  // [64][H][NR][Nn]
                            _Float16* __restrict__ O) {       // [B][NR][192]
  const int h = blockIdx.x, b = blockIdx.y;
  const int w = b & (NWIN - 1);
  __shared__ _Float16 Ks[NR * HD];   // [key][hd]
  __shared__ _Float16 Vt[HD * NR];   // [hd][key] (transposed)
  __shared__ _Float16 Ps[4][16 * NR];

  const _Float16* Qb = Q + ((size_t)(b * Hh + h)) * NR * HD;
  const uint4* K4 = (const uint4*)(K + ((size_t)(b * Hh + h)) * NR * HD);
  const uint4* V4 = (const uint4*)(V + ((size_t)(b * Hh + h)) * NR * HD);

  // vectorized staging: 896 uint4, all rows valid (padded buffers)
  for (int i = threadIdx.x; i < NR * HD / 8; i += blockDim.x) {
    U4H8 vq;
    uint4 kq = K4[i];
    vq.u = V4[i];
    *(uint4*)&Ks[i * 8] = kq;
    int r = i >> 2, c0 = (i & 3) * 8;   // row=key, col base in hd
#pragma unroll
    for (int j = 0; j < 8; ++j) Vt[(c0 + j) * NR + r] = vq.h[j];
  }
  __syncthreads();

  const int lane = threadIdx.x & 31, wv = threadIdx.x >> 5;
  const int ln = lane & 15, half = lane >> 4, kb = half * 8;

  for (int rb = wv; rb < NT; rb += 4) {
    // Q A-frag: 16 rows x K=32 (hd) -> single WMMA per S tile
    v16h a = fragA(Qb + (size_t)(rb * 16 + ln) * HD, kb);

    // combined bias+mask base: element [r][ct] at constant offset r*Nn + ct*16
    const _Float16* cbp = CB + (((size_t)(w * Hh + h) * NR) + rb * 16 + half * 8) * Nn + ln;

    float s[NT][8];
#pragma unroll
    for (int ct = 0; ct < NT; ++ct) {
      const v16h* bk = (const v16h*)(Ks + (ct * 16 + ln) * HD + half * 16);
      v8f acc = {0.f, 0.f, 0.f, 0.f, 0.f, 0.f, 0.f, 0.f};
      acc = wmma16(a, *bk, acc);
#pragma unroll
      for (int r = 0; r < 8; ++r) {
        if (ct < 13) {
          s[ct][r] = acc[r] + (float)cbp[r * Nn + ct * 16];   // const offsets
        } else {
          // last column tile: keys 208..223, only ln<2 are real keys
          s[ct][r] = (ln < Nn - 13 * 16)
                         ? acc[r] + (float)cbp[r * Nn + 13 * 16]
                         : -1e30f;
        }
      }
    }

    // row softmax: each row lives on 16 lanes (same half) x NT register slots
#pragma unroll
    for (int r = 0; r < 8; ++r) {
      float mx = s[0][r];
#pragma unroll
      for (int ct = 1; ct < NT; ++ct) mx = fmaxf(mx, s[ct][r]);
#pragma unroll
      for (int msk = 1; msk < 16; msk <<= 1) mx = fmaxf(mx, __shfl_xor(mx, msk, 32));
      float sum = 0.f;
#pragma unroll
      for (int ct = 0; ct < NT; ++ct) { float e = __expf(s[ct][r] - mx); s[ct][r] = e; sum += e; }
#pragma unroll
      for (int msk = 1; msk < 16; msk <<= 1) sum += __shfl_xor(sum, msk, 32);
      float inv = 1.f / sum;
#pragma unroll
      for (int ct = 0; ct < NT; ++ct) s[ct][r] *= inv;
    }

    // stage P (f16) in per-wave LDS to reshape D-layout -> A-layout
    _Float16* P = Ps[wv];
#pragma unroll
    for (int ct = 0; ct < NT; ++ct)
#pragma unroll
      for (int r = 0; r < 8; ++r)
        P[(half * 8 + r) * NR + ct * 16 + ln] = (_Float16)s[ct][r];

    // P @ V : K-dim = 224 keys in 7 chunks of 32, output 16x32 (two col tiles)
    v8f o0 = {0.f, 0.f, 0.f, 0.f, 0.f, 0.f, 0.f, 0.f};
    v8f o1 = {0.f, 0.f, 0.f, 0.f, 0.f, 0.f, 0.f, 0.f};
#pragma unroll
    for (int kc = 0; kc < 7; ++kc) {
      v16h ap = fragA(P + ln * NR + kc * 32, kb);
      const v16h* b0 = (const v16h*)(Vt + (size_t)ln * NR + kc * 32 + half * 16);
      const v16h* b1 = (const v16h*)(Vt + (size_t)(16 + ln) * NR + kc * 32 + half * 16);
      o0 = wmma16(ap, *b0, o0);
      o1 = wmma16(ap, *b1, o1);
    }
    // padded O rows -> unconditional stores
    _Float16* orow = O + ((size_t)b * NR + rb * 16 + half * 8) * Dm + h * HD;
#pragma unroll
    for (int r = 0; r < 8; ++r) {
      orow[r * Dm + ln]      = (_Float16)o0[r];
      orow[r * Dm + 16 + ln] = (_Float16)o1[r];
    }
  }
}

// ---------------------------------------------------------------------------
// Kernel 3: output projection (f16 in -> f32 d_out), grid (NT, B), 4 waves
// ---------------------------------------------------------------------------
__global__ void proj_kernel(const _Float16* __restrict__ Oin,  // [B][NR][192]
                            const _Float16* __restrict__ Wp,   // packed [12][6][2][16][16]
                            const float* __restrict__ bias,    // [192]
                            float* __restrict__ out) {         // [B][N][192]
  const int rt = blockIdx.x, b = blockIdx.y;
  __shared__ _Float16 X[16 * Dm];
  for (int i = threadIdx.x; i < 16 * Dm / 8; i += blockDim.x) {
    // padded input rows -> unconditional b128 loads
    *(uint4*)&X[i * 8] = *(const uint4*)(Oin + ((size_t)b * NR + rt * 16) * Dm + i * 8);
  }
  __syncthreads();

  const int lane = threadIdx.x & 31, wv = threadIdx.x >> 5;
  const int ln = lane & 15, half = lane >> 4, kb = half * 8;

  for (int tt = 0; tt < 3; ++tt) {
    int tl = wv * 3 + tt;
    int cw = tl * 16;
    __builtin_prefetch(Wp + (size_t)tl * 6 * 512, 0, 1);
    v8f acc = {0.f, 0.f, 0.f, 0.f, 0.f, 0.f, 0.f, 0.f};
#pragma unroll
    for (int kc = 0; kc < 6; ++kc) {
      v16h a = fragA(&X[ln * Dm + kc * 32], kb);
      const v16h* wp = (const v16h*)(Wp + ((((size_t)tl * 6 + kc) * 2 + half) * 16 + ln) * 16);
      acc = wmma16(a, *wp, acc);
    }
    float bb = bias[cw + ln];
    float* orow = out + ((size_t)b * Nn + rt * 16 + half * 8) * Dm + cw + ln;
    if (rt * 16 + 15 < Nn) {
      // wave-uniform fast path: all 16 rows valid
#pragma unroll
      for (int r = 0; r < 8; ++r) orow[r * Dm] = acc[r] + bb;
    } else {
#pragma unroll
      for (int r = 0; r < 8; ++r) {
        int row = rt * 16 + half * 8 + r;
        if (row < Nn) orow[r * Dm] = acc[r] + bb;
      }
    }
  }
}

// ---------------------------------------------------------------------------
extern "C" void kernel_launch(void* const* d_in, const int* in_sizes, int n_in,
                              void* d_out, int out_size, void* d_ws, size_t ws_size,
                              hipStream_t stream) {
  (void)in_sizes; (void)n_in; (void)out_size; (void)ws_size;
  const float* mov  = (const float*)d_in[0];
  const float* fix  = (const float*)d_in[1];
  const float* dmov = (const float*)d_in[2];
  const float* dfix = (const float*)d_in[3];
  const float* mask = (const float*)d_in[4];
  const int*   rel  = (const int*)  d_in[5];
  const float* rpb  = (const float*)d_in[6];
  const float* mqw  = (const float*)d_in[7];
  const float* mqb  = (const float*)d_in[8];
  const float* mkvw = (const float*)d_in[9];
  const float* mkvb = (const float*)d_in[10];
  const float* fqw  = (const float*)d_in[11];
  const float* fqb  = (const float*)d_in[12];
  const float* fkvw = (const float*)d_in[13];
  const float* fkvb = (const float*)d_in[14];
  const float* mpw  = (const float*)d_in[15];
  const float* mpb  = (const float*)d_in[16];
  const float* fpw  = (const float*)d_in[17];
  const float* fpb  = (const float*)d_in[18];

  char* ws = (char*)d_ws;
  const size_t szp = (size_t)Bn * Hh * NR * HD * sizeof(_Float16);  // 22,020,096B
  _Float16* Qm = (_Float16*)(ws + 0 * szp);
  _Float16* Qf = (_Float16*)(ws + 1 * szp);
  _Float16* Km = (_Float16*)(ws + 2 * szp);
  _Float16* Vm = (_Float16*)(ws + 3 * szp);
  _Float16* Kf = (_Float16*)(ws + 4 * szp);
  _Float16* Vf = (_Float16*)(ws + 5 * szp);
  _Float16* Om = (_Float16*)(ws + 6 * szp);   // [B][NR][Dm] (same bytes)
  _Float16* Of = (_Float16*)(ws + 7 * szp);
  float* biasws = (float*)(ws + 8 * szp);            // 1,058,400B (padded 1,058,560)
  char* pb = ws + 8 * szp + 1058560;
  _Float16* Pmq  = (_Float16*)(pb);                  //  73,728B
  _Float16* Pfq  = (_Float16*)(pb + 73728);          //  73,728B
  _Float16* Pmkv = (_Float16*)(pb + 147456);         // 147,456B
  _Float16* Pfkv = (_Float16*)(pb + 294912);         // 147,456B
  _Float16* Pmp  = (_Float16*)(pb + 442368);         //  73,728B
  _Float16* Pfp  = (_Float16*)(pb + 516096);         //  73,728B
  _Float16* CB   = (_Float16*)(pb + 589824);         // 36,126,720B

  bias_kernel<<<(Hh * Nn * Nn + 255) / 256, 256, 0, stream>>>(rel, rpb, biasws);
  comb_kernel<<<(NWIN * Hh * NR * Nn + 255) / 256, 256, 0, stream>>>(biasws, mask, CB);
  pack_w_kernel<<<(Dm * Dm + 255) / 256, 256, 0, stream>>>(mqw, Dm, Pmq);
  pack_w_kernel<<<(Dm * Dm + 255) / 256, 256, 0, stream>>>(fqw, Dm, Pfq);
  pack_w_kernel<<<(Dm * 2 * Dm + 255) / 256, 256, 0, stream>>>(mkvw, 2 * Dm, Pmkv);
  pack_w_kernel<<<(Dm * 2 * Dm + 255) / 256, 256, 0, stream>>>(fkvw, 2 * Dm, Pfkv);
  pack_w_kernel<<<(Dm * Dm + 255) / 256, 256, 0, stream>>>(mpw, Dm, Pmp);
  pack_w_kernel<<<(Dm * Dm + 255) / 256, 256, 0, stream>>>(fpw, Dm, Pfp);

  qkv_kernel<<<dim3(NT, Bn), 256, 0, stream>>>(mov, fix, dmov, dfix,
                                               Pmq, mqb, Pmkv, mkvb,
                                               Pfq, fqb, Pfkv, fkvb,
                                               Qm, Qf, Km, Vm, Kf, Vf);

  // mov branch: fix_q x dmov_k -> dmov_v ; fix branch: mov_q x dfix_k -> dfix_v
  attn_kernel<<<dim3(Hh, Bn), 128, 0, stream>>>(Qf, Km, Vm, CB, Om);
  attn_kernel<<<dim3(Hh, Bn), 128, 0, stream>>>(Qm, Kf, Vf, CB, Of);

  float* out = (float*)d_out;
  proj_kernel<<<dim3(NT, Bn), 128, 0, stream>>>(Om, Pmp, mpb, out);
  proj_kernel<<<dim3(NT, Bn), 128, 0, stream>>>(Of, Pfp, fpb, out + (size_t)Bn * Nn * Dm);
}
